// RBFSetConv_45930380263686
// MI455X (gfx1250) — matrix operand order repaired and checked by reading, not compiled, not run on hardware
//
#include <hip/hip_runtime.h>

// ---------------------------------------------------------------------------
// RBF set-convolution onto a 64x64 grid, CDNA5 (gfx1250, wave32, WMMA).
//
// out[b,0,g] = density = sum_n w[b,n,g]
// out[b,1,g] = sum_n y w / (density+eps),  out[b,2,g] = sum_n t w / (...)
// w = exp(-0.5 |x_n - grid_g|^2 / sigma^2) * !mask
//
// Reduction over n is the GEMM [1;y;t](3xN pad 16) x W(NxG) on the XDL via
// V_WMMA_F32_16X16X32_F16 (two independent accumulators -> 2 WMMAs in
// flight). Weight generation is reduced to add+2fma+v_exp_f32+cvt per pair:
// exponent = C0(g) + q(n) + gx*u(n) + gy*v(n), with q,u,v precomputed per
// point at staging time and log2(e) folded into the sigma scale so the
// exponential is a bare exp2.
// ---------------------------------------------------------------------------

#define GRID_RES 64
constexpr int   G_TOT  = GRID_RES * GRID_RES;   // 4096 grid points
constexpr int   NPTS   = 4096;                  // points per batch
constexpr int   BATCH  = 8;
constexpr float SIGMA  = 0.1f;
constexpr float EPS_D  = 1e-5f;
constexpr float LOG2E  = 1.4426950408889634f;

typedef __attribute__((ext_vector_type(16))) _Float16 v16h;
typedef __attribute__((ext_vector_type(8)))  _Float16 v8h;
typedef __attribute__((ext_vector_type(8)))  float    v8f;
typedef __attribute__((ext_vector_type(2)))  float    v2f;

// A tile (feats, 16x32 f16). Lane supplies row M = col of [1; y; t; 0...].
// Lanes 0-15 hold K = 0..15; lanes 16-31 hold K = 16..31.
__device__ __forceinline__ v16h build_feats_tile(const _Float16* yrow,
                                                 const _Float16* trow,
                                                 int half, int col)
{
    const v8h ylo = *(const v8h*)(yrow + 16 * half);
    const v8h yhi = *(const v8h*)(yrow + 16 * half + 8);
    const v8h tlo = *(const v8h*)(trow + 16 * half);
    const v8h thi = *(const v8h*)(trow + 16 * half + 8);
    v16h a;
    #pragma unroll
    for (int i = 0; i < 16; ++i) {
        const _Float16 vy = (i < 8) ? ylo[i] : yhi[i - 8];
        const _Float16 vt = (i < 8) ? tlo[i] : thi[i - 8];
        _Float16 val = (_Float16)0.0f;           // rows 3..15: zero padding
        if (col == 0) val = (_Float16)1.0f;      // row 0: ones -> density
        if (col == 1) val = vy;                  // row 1: y feature
        if (col == 2) val = vt;                  // row 2: t feature
        a[i] = val;
    }
    return a;
}

// B tile (weights, 32x16 f16). Lane owns column g; lanes 0-15: K=0..15,
// lanes 16-31: K=16..31, packed consecutively. 3 VALU + exp2 + cvt per pair.
__device__ __forceinline__ v16h build_weight_tile(const float4* prow, int kb,
                                                  float gx, float gy, float c0)
{
    v16h bm;
    #pragma unroll
    for (int j = 0; j < 16; ++j) {
        const float4 p = prow[kb + j];      // {u, v, q, 0} -> one ds_load_b128
        float e = c0 + p.z;
        e = fmaf(gy, p.y, e);
        e = fmaf(gx, p.x, e);
        bm[j] = (_Float16)__builtin_amdgcn_exp2f(e);   // bare v_exp_f32
    }
    return bm;
}

__global__ __launch_bounds__(128)
void rbf_setconv_wmma(const float* __restrict__ x_c,        // (B, N, 2)
                      const float* __restrict__ y_c,        // (B, N, 1)
                      const float* __restrict__ t_c,        // (B, N, 1)
                      const unsigned char* __restrict__ msk,// (B, N) bool
                      float* __restrict__ out)              // (B, 3, 64, 64)
{
    // Per-wave private staging (no cross-wave sharing -> no barriers; LDS ops
    // from one wave complete in order, pinned by s_wait_dscnt below).
    __shared__ __align__(16) float4   s_p[4][64];   // {u, v, q, 0} per point
    __shared__ __align__(16) _Float16 s_y[4][64];
    __shared__ __align__(16) _Float16 s_t[4][64];

    const int tid  = threadIdx.x;
    const int lane = tid & 31;          // wave32 lane
    const int wave = tid >> 5;          // 4 waves / block
    const int col  = lane & 15;         // N column of the 16x16 tile
    const int half = lane >> 4;         // K-half selector

    const int b  = blockIdx.x >> 6;                  // batch
    const int g0 = ((blockIdx.x & 63) << 6) + wave * 16;
    const int g  = g0 + col;                         // this lane's grid point

    // grid[g] = (coords[g % 64], coords[g / 64]), coords[j] = -1 + 2j/63
    const float gx = -1.0f + (2.0f / 63.0f) * (float)(g & 63);
    const float gy = -1.0f + (2.0f / 63.0f) * (float)(g >> 6);

    const float s2 = (-0.5f / (SIGMA * SIGMA)) * LOG2E;  // base-2 exponent scale
    const float m2 = -2.0f * s2;
    const float c0 = s2 * (gx * gx + gy * gy);           // per-lane constant

    const float*         xb = x_c + (size_t)b * NPTS * 2;
    const float*         yb = y_c + (size_t)b * NPTS;
    const float*         tb = t_c + (size_t)b * NPTS;
    const unsigned char* mb = msk + (size_t)b * NPTS;

    v8f acc0 = {};   // two independent C tiles -> two WMMAs in flight
    v8f acc1 = {};

    for (int n0 = 0; n0 < NPTS; n0 += 64) {
        // ---- stage 64 points: lane L loads n0+L and n0+32+L ----------------
        #pragma unroll
        for (int c = 0; c < 2; ++c) {
            const int n = n0 + 32 * c + lane;
            const v2f   xv = *(const v2f*)(xb + 2 * n);
            const float yv = yb[n];
            const float tv = tb[n];
            float q = s2 * fmaf(xv.x, xv.x, xv.y * xv.y);
            if (mb[n]) q = -3.0e38f;                 // masked -> exp2 == 0
            s_p[wave][32 * c + lane] = make_float4(m2 * xv.x, m2 * xv.y, q, 0.0f);
            s_y[wave][32 * c + lane] = (_Float16)yv;
            s_t[wave][32 * c + lane] = (_Float16)tv;
        }
        if (n0 + 64 < NPTS)                          // pull next chunk to L0
            __builtin_prefetch(xb + 2 * (n0 + 64 + lane), 0, 0);

        asm volatile("s_wait_dscnt 0" ::: "memory"); // DS store->load ordering

        // ---- build tiles and accumulate on the XDL -------------------------
        const v16h a0 = build_feats_tile(&s_y[wave][0],  &s_t[wave][0],  half, col);
        const v16h a1 = build_feats_tile(&s_y[wave][32], &s_t[wave][32], half, col);
        const v16h b0 = build_weight_tile(&s_p[wave][0],  16 * half, gx, gy, c0);
        const v16h b1 = build_weight_tile(&s_p[wave][32], 16 * half, gx, gy, c0);

        acc0 = __builtin_amdgcn_wmma_f32_16x16x32_f16(
            false, a0, false, b0, (short)0, acc0, false, false);
        acc1 = __builtin_amdgcn_wmma_f32_16x16x32_f16(
            false, a1, false, b1, (short)0, acc1, false, false);
    }

    // ---- epilogue: lanes 0-15 hold rows M=0..7 of the C tile ---------------
    if (half == 0) {
        const float dens = acc0[0] + acc1[0];        // M=0: density
        const float wy   = acc0[1] + acc1[1];        // M=1: sum(y*w)
        const float wt   = acc0[2] + acc1[2];        // M=2: sum(t*w)
        const float inv  = 1.0f / (dens + EPS_D);
        float* ob = out + (size_t)b * 3 * G_TOT;
        ob[g]             = dens;
        ob[G_TOT + g]     = wy * inv;
        ob[2 * G_TOT + g] = wt * inv;
    }
}

extern "C" void kernel_launch(void* const* d_in, const int* in_sizes, int n_in,
                              void* d_out, int out_size, void* d_ws, size_t ws_size,
                              hipStream_t stream) {
    const float*         x_c  = (const float*)d_in[0];        // (8,4096,2) f32
    const float*         y_c  = (const float*)d_in[1];        // (8,4096,1) f32
    const float*         t_c  = (const float*)d_in[2];        // (8,4096,1) f32
    const unsigned char* mask = (const unsigned char*)d_in[3];// (8,4096) bool (1B)
    float*               out  = (float*)d_out;                // (8,3,64,64) f32

    // one block = 4 waves = 64 grid points; 8 batches * 64 blocks-per-batch
    dim3 grid(BATCH * (G_TOT / 64));
    dim3 block(128);
    hipLaunchKernelGGL(rbf_setconv_wmma, grid, block, 0, stream,
                       x_c, y_c, t_c, mask, out);
}
